// VGG_SMALL_bi_1W1A_65068754534654
// MI455X (gfx1250) — 8-bit matrix operand order repaired and checked by reading, compiled, not run on hardware
//
#include <hip/hip_runtime.h>
#include <hip/hip_bf16.h>
#include <math.h>


typedef int v8i __attribute__((ext_vector_type(8)));

__device__ __forceinline__ unsigned ld_u32(const signed char* p) {
  return *reinterpret_cast<const unsigned*>(p);
}

__device__ __forceinline__ float block_reduce(float v, float* red) {
  const int tid = threadIdx.x;
  red[tid] = v;
  __syncthreads();
  for (int st = blockDim.x >> 1; st > 0; st >>= 1) {
    if (tid < st) red[tid] += red[tid + st];
    __syncthreads();
  }
  float r = red[0];
  __syncthreads();
  return r;
}

__global__ void bits_kernel(const float* __restrict__ x,
                            signed char* __restrict__ acol, int Bn) {
  const long Npix = (long)Bn * 32 * 32;
  const long pix = (long)blockIdx.x * blockDim.x + threadIdx.x;
  if (pix >= Npix) return;
  const int b = (int)(pix >> 10);
  const int hw = (int)(pix & 1023);
  const int oh = hw >> 5, ow = hw & 31;

  __align__(16) signed char tmp[64];
#pragma unroll
  for (int i = 0; i < 64; ++i) tmp[i] = 0;

  for (int ic = 0; ic < 3; ++ic) {
    const float origin = x[((b * 3 + ic) * 32 + oh) * 32 + ow] * 255.0f;
    float p = 128.0f;
#pragma unroll
    for (int j = 0; j < 7; ++j) {
      const float bit = floorf(fmodf(origin, 2.0f * p) / p);
      tmp[ic * 8 + j] = (signed char)(bit > 0.5f ? 1 : -1);
      p *= 0.5f;
    }
    const float v = (fmodf(origin, 2.0f) - 0.5f) * 2.0f;
    tmp[ic * 8 + 7] = (signed char)((v > 0.0f) - (v < 0.0f));
  }
  int4* dst = reinterpret_cast<int4*>(acol + pix * 64);
  const int4* src = reinterpret_cast<const int4*>(tmp);
  dst[0] = src[0]; dst[1] = src[1]; dst[2] = src[2]; dst[3] = src[3];
}

__global__ void binw3x3_kernel(const float* __restrict__ w,
                               signed char* __restrict__ wt,
                               float* __restrict__ sw, int OC, int IC) {
  __shared__ float red[256];
  const int oc = blockIdx.x;
  const int n = IC * 9;
  const int tid = threadIdx.x;
  const float* wp = w + (long)oc * n;

  float s = 0.f;
  for (int i = tid; i < n; i += blockDim.x) s += wp[i];
  const float mean = block_reduce(s, red) / (float)n;

  float s2 = 0.f;
  for (int i = tid; i < n; i += blockDim.x) {
    const float d = wp[i] - mean;
    s2 += d * d;
  }
  const float stdv = sqrtf(block_reduce(s2, red) / (float)(n - 1));

  float sa = 0.f;
  for (int i = tid; i < n; i += blockDim.x) sa += fabsf((wp[i] - mean) / stdv);
  const float mabs = block_reduce(sa, red) / (float)n;
  const float scale = exp2f(rintf(log2f(mabs)));
  if (tid == 0) sw[oc] = scale;

  for (int i = tid; i < n; i += blockDim.x) {
    const float bw = wp[i] - mean;
    const int ic = i / 9, t = i % 9;
    wt[((long)t * OC + oc) * IC + ic] = (signed char)((bw > 0.f) - (bw < 0.f));
  }
}

__global__ void binw0_kernel(const float* __restrict__ w,
                             signed char* __restrict__ wt,
                             float* __restrict__ sw) {
  __shared__ float red[256];
  const int oc = blockIdx.x;
  const int n = 24;
  const int tid = threadIdx.x;
  const float* wp = w + (long)oc * n;

  float s = 0.f;
  for (int i = tid; i < n; i += blockDim.x) s += wp[i];
  const float mean = block_reduce(s, red) / (float)n;

  float s2 = 0.f;
  for (int i = tid; i < n; i += blockDim.x) {
    const float d = wp[i] - mean;
    s2 += d * d;
  }
  const float stdv = sqrtf(block_reduce(s2, red) / (float)(n - 1));

  float sa = 0.f;
  for (int i = tid; i < n; i += blockDim.x) sa += fabsf((wp[i] - mean) / stdv);
  const float mabs = block_reduce(sa, red) / (float)n;
  if (tid == 0) sw[oc] = exp2f(rintf(log2f(mabs)));

  for (int k = tid; k < 64; k += blockDim.x) {
    signed char v = 0;
    if (k < 24) {
      const float bw = wp[k] - mean;
      v = (signed char)((bw > 0.f) - (bw < 0.f));
    }
    wt[(long)oc * 64 + k] = v;
  }
}

__global__ void conv0_wmma(const signed char* __restrict__ acol,
                           const signed char* __restrict__ wt,
                           short* __restrict__ out,
                           int Npix) {
  const int lane = threadIdx.x & 31;
  const int gwave = blockIdx.x * (blockDim.x >> 5) + (threadIdx.x >> 5);
  const int pixTiles = Npix >> 4;
  if (gwave >= pixTiles) return;

  const int n = lane & 15;
  const bool hi = lane >= 16;
  const long pix = (long)gwave * 16 + n;
  const signed char* ap = acol + pix * 64;

  v8i Bv;
#pragma unroll
  for (int r = 0; r < 8; ++r) {
    const int kb = (r >> 2) * 32 + (r & 3) * 4 + (hi ? 16 : 0);
    Bv[r] = (int)ld_u32(ap + kb);
  }

  v8i acc[8];
#pragma unroll
  for (int t = 0; t < 8; ++t) {
    const signed char* wp = wt + (long)(t * 16 + n) * 64;
    v8i A;
#pragma unroll
    for (int r = 0; r < 8; ++r) {
      const int ka = (r >> 2) * 32 + (r & 3) * 4 + (hi ? 16 : 0);
      A[r] = (int)ld_u32(wp + ka);
    }
    v8i z = {};
    acc[t] = __builtin_amdgcn_wmma_i32_16x16x64_iu8(true, A, true, Bv, z,
                                                    false, false);
  }

#pragma unroll
  for (int t = 0; t < 8; ++t) {
    __align__(16) short res[8];
#pragma unroll
    for (int r = 0; r < 8; ++r) res[r] = (short)acc[t][r];
    *reinterpret_cast<int4*>(out + pix * 128 + t * 16 + (hi ? 8 : 0)) =
        *reinterpret_cast<const int4*>(res);
  }
}

#define OCT 4
__global__ void conv3x3_wmma(const signed char* __restrict__ act,
                             const signed char* __restrict__ wt,
                             short* __restrict__ out,
                             int Bn, int H, int W, int IC, int OC) {
  const int lane = threadIdx.x & 31;
  const int gwave = blockIdx.x * (blockDim.x >> 5) + (threadIdx.x >> 5);
  const int Npix = Bn * H * W;
  const int pixTiles = Npix >> 4;
  const int ocGroups = OC / (16 * OCT);
  if (gwave >= pixTiles * ocGroups) return;
  const int pixTile = gwave % pixTiles;
  const int ocGroup = gwave / pixTiles;

  const int n = lane & 15;
  const bool hi = lane >= 16;
  const long pix = (long)pixTile * 16 + n;
  const int b = (int)(pix / (H * W));
  const int hw = (int)(pix % (H * W));
  const int oh = hw / W, ow = hw % W;
  const int ocBase = ocGroup * (16 * OCT);

  v8i acc[OCT];
#pragma unroll
  for (int t = 0; t < OCT; ++t) acc[t] = (v8i){};

  for (int tap = 0; tap < 9; ++tap) {
    const int kh = tap / 3, kw = tap % 3;
    const int ih = oh + kh - 1, iw = ow + kw - 1;
    const bool inb = (ih >= 0) & (ih < H) & (iw >= 0) & (iw < W);
    const signed char* actp = act + (((long)b * H + ih) * W + iw) * IC;
    const signed char* wtp0 = wt + ((long)tap * OC + ocBase + n) * IC;
    for (int kk = 0; kk < IC; kk += 64) {
      v8i Bv;
#pragma unroll
      for (int r = 0; r < 8; ++r) {
        const int kb = kk + (r >> 2) * 32 + (r & 3) * 4 + (hi ? 16 : 0);
        unsigned bv = 0u;
        if (inb) bv = ld_u32(actp + kb);
        Bv[r] = (int)bv;
      }
#pragma unroll
      for (int t = 0; t < OCT; ++t) {
        const signed char* wtp = wtp0 + (long)(t * 16) * IC;
        v8i A;
#pragma unroll
        for (int r = 0; r < 8; ++r) {
          const int ka = kk + (r >> 2) * 32 + (r & 3) * 4 + (hi ? 16 : 0);
          A[r] = (int)ld_u32(wtp + ka);
        }
        acc[t] = __builtin_amdgcn_wmma_i32_16x16x64_iu8(true, A, true, Bv,
                                                        acc[t], false, false);
      }
    }
  }

#pragma unroll
  for (int t = 0; t < OCT; ++t) {
    __align__(16) short res[8];
#pragma unroll
    for (int r = 0; r < 8; ++r) res[r] = (short)acc[t][r];
    *reinterpret_cast<int4*>(out + pix * OC + ocBase + t * 16 + (hi ? 8 : 0)) =
        *reinterpret_cast<const int4*>(res);
  }
}

__global__ void pool2_kernel(const short* __restrict__ in,
                             short* __restrict__ out,
                             int Bn, int H, int W, int C) {
  const long total = (long)Bn * (H / 2) * (W / 2) * C;
  const long idx = (long)blockIdx.x * blockDim.x + threadIdx.x;
  if (idx >= total) return;
  const int c = (int)(idx % C);
  long r = idx / C;
  const int ow = (int)(r % (W / 2)); r /= (W / 2);
  const int oh = (int)(r % (H / 2));
  const int b = (int)(r / (H / 2));
  const short* p = in + (((long)b * H + 2 * oh) * W + 2 * ow) * C + c;
  short m = p[0];
  short v = p[C];                if (v > m) m = v;
  v = p[(long)W * C];            if (v > m) m = v;
  v = p[(long)W * C + C];        if (v > m) m = v;
  out[idx] = m;
}

__global__ void stats_partial(const short* __restrict__ t,
                              const float* __restrict__ sw,
                              float* __restrict__ partial,
                              long Npix, int C) {
  const int c = threadIdx.x;
  const float sc = sw[c];
  float s = 0.f, s2 = 0.f;
  for (long p = blockIdx.x; p < Npix; p += gridDim.x) {
    const float y = sc * (float)t[p * C + c];
    s += y;
    s2 += y * y;
  }
  const long base = ((long)blockIdx.x * C + c) * 2;
  partial[base] = s;
  partial[base + 1] = s2;
}

__global__ void stats_final(const float* __restrict__ partial,
                            float* __restrict__ stats,
                            int NB, int C) {
  __shared__ float red[256];
  const int c = blockIdx.x;
  const int tid = threadIdx.x;
  float s = 0.f, s2 = 0.f;
  for (int i = tid; i < NB; i += blockDim.x) {
    const long base = ((long)i * C + c) * 2;
    s += partial[base];
    s2 += partial[base + 1];
  }
  const float ts = block_reduce(s, red);
  const float ts2 = block_reduce(s2, red);
  if (tid == 0) { stats[c * 2] = ts; stats[c * 2 + 1] = ts2; }
}

__global__ void bn_out_kernel(const short* __restrict__ t,
                              const float* __restrict__ sw,
                              const float* __restrict__ stats,
                              const float* __restrict__ g,
                              const float* __restrict__ bt,
                              signed char* __restrict__ outa,
                              float* __restrict__ outf,
                              long Npix, int C, int mode) {
  const long total = Npix * C;
  const long idx = (long)blockIdx.x * blockDim.x + threadIdx.x;
  if (idx >= total) return;
  const int c = (int)(idx % C);
  const float N = (float)Npix;
  const float m = stats[c * 2] / N;
  const float var = stats[c * 2 + 1] / N - m * m;
  const float y = sw[c] * (float)t[idx];
  const float bnv = (y - m) * rsqrtf(var + 1e-5f) * g[c] + bt[c];
  if (mode == 0) {
    outa[idx] = (signed char)((bnv > 0.f) - (bnv < 0.f));
  } else {
    outf[idx] = fminf(1.0f, fmaxf(-1.0f, bnv));
  }
}

__global__ void fc_kernel(const float* __restrict__ h,
                          const float* __restrict__ fw,
                          const float* __restrict__ fb,
                          float* __restrict__ out) {
  __shared__ float red[256];
  const int b = blockIdx.x;
  const int o = blockIdx.y;
  float s = 0.f;
  for (int k = threadIdx.x; k < 8192; k += blockDim.x) {
    const int c = k >> 4, hw = k & 15;
    s += h[((long)b * 16 + hw) * 512 + c] * fw[(long)o * 8192 + k];
  }
  const float ts = block_reduce(s, red);
  if (threadIdx.x == 0) out[b * 10 + o] = ts + fb[o];
}

static inline size_t alignup(size_t x) { return (x + 255) & ~(size_t)255; }

extern "C" void kernel_launch(void* const* d_in, const int* in_sizes, int n_in,
                              void* d_out, int out_size, void* d_ws,
                              size_t ws_size, hipStream_t stream) {
  (void)in_sizes; (void)n_in; (void)out_size; (void)ws_size;

  const float* x = (const float*)d_in[0];
  const float* w[6] = {(const float*)d_in[1], (const float*)d_in[2],
                       (const float*)d_in[3], (const float*)d_in[4],
                       (const float*)d_in[5], (const float*)d_in[6]};
  const float* fcw = (const float*)d_in[7];
  const float* fcb = (const float*)d_in[8];
  const float* g[6]; const float* bt[6];
  for (int i = 0; i < 6; ++i) { g[i] = (const float*)d_in[9 + i]; bt[i] = (const float*)d_in[15 + i]; }
  float* out = (float*)d_out;

  char* ws = (char*)d_ws;
  size_t off = 0;
  auto take = [&](size_t bytes) { char* p = ws + off; off = alignup(off + bytes); return p; };

  signed char* a0col = (signed char*)take(262144ull * 64);
  signed char* wt[6];
  wt[0] = (signed char*)take(128ull * 64);
  wt[1] = (signed char*)take(9ull * 128 * 128);
  wt[2] = (signed char*)take(9ull * 256 * 128);
  wt[3] = (signed char*)take(9ull * 256 * 256);
  wt[4] = (signed char*)take(9ull * 512 * 256);
  wt[5] = (signed char*)take(9ull * 512 * 512);
  float* sw[6];
  for (int i = 0; i < 6; ++i) sw[i] = (float*)take(512 * 4);
  signed char* actA = (signed char*)take(33554432ull);
  signed char* actB = (signed char*)take(33554432ull);
  short* convb = (short*)take(67108864ull);
  short* poolb = (short*)take(16777216ull);
  float* partial = (float*)take(256ull * 512 * 2 * 4);
  float* stats = (float*)take(512ull * 2 * 4);
  float* hbuf = (float*)take(4096ull * 512 * 4);

  const int Bn = 256;
  const int NB = 256;

  bits_kernel<<<(262144 + 255) / 256, 256, 0, stream>>>(x, a0col, Bn);
  binw0_kernel<<<128, 256, 0, stream>>>(w[0], wt[0], sw[0]);
  binw3x3_kernel<<<128, 256, 0, stream>>>(w[1], wt[1], sw[1], 128, 128);
  binw3x3_kernel<<<256, 256, 0, stream>>>(w[2], wt[2], sw[2], 256, 128);
  binw3x3_kernel<<<256, 256, 0, stream>>>(w[3], wt[3], sw[3], 256, 256);
  binw3x3_kernel<<<512, 256, 0, stream>>>(w[4], wt[4], sw[4], 512, 256);
  binw3x3_kernel<<<512, 256, 0, stream>>>(w[5], wt[5], sw[5], 512, 512);

  {
    const int Npix = 262144, C = 128;
    conv0_wmma<<<(Npix / 16) / 4, 128, 0, stream>>>(a0col, wt[0], convb, Npix);
    stats_partial<<<NB, C, 0, stream>>>(convb, sw[0], partial, Npix, C);
    stats_final<<<C, 256, 0, stream>>>(partial, stats, NB, C);
    const long tot = (long)Npix * C;
    bn_out_kernel<<<(int)((tot + 255) / 256), 256, 0, stream>>>(
        convb, sw[0], stats, g[0], bt[0], actA, nullptr, Npix, C, 0);
  }
  {
    const int H = 32, W = 32, IC = 128, OC = 128;
    const int Npix = Bn * H * W;
    const int nT = (Npix / 16) * (OC / (16 * OCT));
    conv3x3_wmma<<<nT / 4, 128, 0, stream>>>(actA, wt[1], convb, Bn, H, W, IC, OC);
    const long ptot = (long)Bn * (H / 2) * (W / 2) * OC;
    pool2_kernel<<<(int)((ptot + 255) / 256), 256, 0, stream>>>(convb, poolb, Bn, H, W, OC);
    const long Np = (long)Bn * 16 * 16;
    stats_partial<<<NB, OC, 0, stream>>>(poolb, sw[1], partial, Np, OC);
    stats_final<<<OC, 256, 0, stream>>>(partial, stats, NB, OC);
    bn_out_kernel<<<(int)((Np * OC + 255) / 256), 256, 0, stream>>>(
        poolb, sw[1], stats, g[1], bt[1], actB, nullptr, Np, OC, 0);
  }
  {
    const int H = 16, W = 16, IC = 128, OC = 256;
    const int Npix = Bn * H * W;
    const int nT = (Npix / 16) * (OC / (16 * OCT));
    conv3x3_wmma<<<nT / 4, 128, 0, stream>>>(actB, wt[2], convb, Bn, H, W, IC, OC);
    stats_partial<<<NB, OC, 0, stream>>>(convb, sw[2], partial, (long)Npix, OC);
    stats_final<<<OC, 256, 0, stream>>>(partial, stats, NB, OC);
    bn_out_kernel<<<(int)(((long)Npix * OC + 255) / 256), 256, 0, stream>>>(
        convb, sw[2], stats, g[2], bt[2], actA, nullptr, (long)Npix, OC, 0);
  }
  {
    const int H = 16, W = 16, IC = 256, OC = 256;
    const int Npix = Bn * H * W;
    const int nT = (Npix / 16) * (OC / (16 * OCT));
    conv3x3_wmma<<<nT / 4, 128, 0, stream>>>(actA, wt[3], convb, Bn, H, W, IC, OC);
    const long ptot = (long)Bn * 8 * 8 * OC;
    pool2_kernel<<<(int)((ptot + 255) / 256), 256, 0, stream>>>(convb, poolb, Bn, H, W, OC);
    const long Np = (long)Bn * 8 * 8;
    stats_partial<<<NB, OC, 0, stream>>>(poolb, sw[3], partial, Np, OC);
    stats_final<<<OC, 256, 0, stream>>>(partial, stats, NB, OC);
    bn_out_kernel<<<(int)((Np * OC + 255) / 256), 256, 0, stream>>>(
        poolb, sw[3], stats, g[3], bt[3], actB, nullptr, Np, OC, 0);
  }
  {
    const int H = 8, W = 8, IC = 256, OC = 512;
    const int Npix = Bn * H * W;
    const int nT = (Npix / 16) * (OC / (16 * OCT));
    conv3x3_wmma<<<nT / 4, 128, 0, stream>>>(actB, wt[4], convb, Bn, H, W, IC, OC);
    stats_partial<<<NB, OC, 0, stream>>>(convb, sw[4], partial, (long)Npix, OC);
    stats_final<<<OC, 256, 0, stream>>>(partial, stats, NB, OC);
    bn_out_kernel<<<(int)(((long)Npix * OC + 255) / 256), 256, 0, stream>>>(
        convb, sw[4], stats, g[4], bt[4], actA, nullptr, (long)Npix, OC, 0);
  }
  {
    const int H = 8, W = 8, IC = 512, OC = 512;
    const int Npix = Bn * H * W;
    const int nT = (Npix / 16) * (OC / (16 * OCT));
    conv3x3_wmma<<<nT / 4, 128, 0, stream>>>(actA, wt[5], convb, Bn, H, W, IC, OC);
    const long ptot = (long)Bn * 4 * 4 * OC;
    pool2_kernel<<<(int)((ptot + 255) / 256), 256, 0, stream>>>(convb, poolb, Bn, H, W, OC);
    const long Np = (long)Bn * 4 * 4;
    stats_partial<<<NB, OC, 0, stream>>>(poolb, sw[5], partial, Np, OC);
    stats_final<<<OC, 256, 0, stream>>>(partial, stats, NB, OC);
    bn_out_kernel<<<(int)((Np * OC + 255) / 256), 256, 0, stream>>>(
        poolb, sw[5], stats, g[5], bt[5], nullptr, hbuf, Np, OC, 1);
  }
  fc_kernel<<<dim3(Bn, 10), 256, 0, stream>>>(hbuf, fcw, fcb, out);
}
